// HungarianMatcher_44882408243449
// MI455X (gfx1250) — matrix operand order repaired and checked by reading, not compiled, not run on hardware
//
#include <hip/hip_runtime.h>
#include <math.h>

// ---------------------------------------------------------------------------
// Hungarian matcher cost matrix on MI455X (gfx1250, wave32, WMMA).
//
//   cost_mask[q,n] = (spSum[q] - <x_q, t_n>) / P
//   cost_dice[q,n] = 1 - (2*<sig_q, t_n> + 1) / (sigSum[q] + tSum[n] + 1)
//   cost_class[q,n] = -softmax(logits_q)[tgt_ids[n]]
//   out = 2*cls + 5*mask + 5*dice
//
// Heavy math: two 1200 x 12544 x 200 GEMMs -> v_wmma_f32_16x16x32_f16,
// fp16 operands (t is exactly representable; x,sigma rounded to fp16), f32 acc.
// Staged fp16 operands (~65 MB) are L2-resident (192 MB L2).
// ---------------------------------------------------------------------------

typedef __attribute__((ext_vector_type(16))) _Float16 v16h;
typedef __attribute__((ext_vector_type(8)))  _Float16 v8h;
typedef __attribute__((ext_vector_type(8)))  float    v8f;

#define NB    4
#define NQ    300
#define BQ    1200          // NB*NQ
#define NC    80
#define HW    65536         // 256*256
#define NT    200           // num targets
#define NTPAD 208           // padded to 13 * 16
#define NP    12544         // sampled points (multiple of 32)
#define KSTEPS (NP / 32)    // 392

// ---------------------------------------------------------------- targets ---
__global__ __launch_bounds__(256)
void prep_targets(const float* __restrict__ tgt_masks,
                  const int*   __restrict__ point_idx,
                  _Float16*    __restrict__ t_h,
                  float*       __restrict__ tSum) {
    const int n   = blockIdx.x;      // 0..NTPAD-1
    const int tid = threadIdx.x;
    __shared__ float red[256];
    float acc = 0.f;
    if (n < NT) {
        const float* row = tgt_masks + (size_t)n * HW;
        for (int p = tid; p < NP; p += 256) {
            float v = row[point_idx[p]];
            t_h[(size_t)n * NP + p] = (_Float16)v;
            acc += v;
        }
    } else {
        for (int p = tid; p < NP; p += 256)
            t_h[(size_t)n * NP + p] = (_Float16)0.f;   // zero pad rows 200..207
    }
    red[tid] = acc;
    __syncthreads();
    for (int s = 128; s > 0; s >>= 1) {
        if (tid < s) red[tid] += red[tid + s];
        __syncthreads();
    }
    if (tid == 0) tSum[n] = red[0];
}

// ---------------------------------------------------------------- queries ---
__global__ __launch_bounds__(256)
void prep_queries(const float* __restrict__ pred_masks,
                  const int*   __restrict__ point_idx,
                  _Float16*    __restrict__ x_h,
                  _Float16*    __restrict__ s_h,
                  float*       __restrict__ spSum,
                  float*       __restrict__ sigSum) {
    const int q   = blockIdx.x;      // 0..BQ-1
    const int tid = threadIdx.x;
    __shared__ float red[256];
    const float* row = pred_masks + (size_t)q * HW;
    float accSp = 0.f, accSig = 0.f;
    for (int p = tid; p < NP; p += 256) {
        float x  = row[point_idx[p]];
        float sg = 1.f / (1.f + __expf(-x));
        float sp = fmaxf(x, 0.f) + log1pf(__expf(-fabsf(x)));  // stable softplus
        x_h[(size_t)q * NP + p] = (_Float16)x;
        s_h[(size_t)q * NP + p] = (_Float16)sg;
        accSp  += sp;
        accSig += sg;
    }
    red[tid] = accSp;  __syncthreads();
    for (int s = 128; s > 0; s >>= 1) { if (tid < s) red[tid] += red[tid + s]; __syncthreads(); }
    if (tid == 0) spSum[q] = red[0];
    __syncthreads();
    red[tid] = accSig; __syncthreads();
    for (int s = 128; s > 0; s >>= 1) { if (tid < s) red[tid] += red[tid + s]; __syncthreads(); }
    if (tid == 0) sigSum[q] = red[0];
}

// -------------------------------------------------------------- class cost --
__global__ __launch_bounds__(128)
void class_prob(const float* __restrict__ pred_logits,
                float*       __restrict__ prob) {
    const int q   = blockIdx.x;      // 0..BQ-1
    const int tid = threadIdx.x;
    __shared__ float red[128];
    float v = (tid < NC) ? pred_logits[(size_t)q * NC + tid] : -1e30f;
    red[tid] = v; __syncthreads();
    for (int s = 64; s > 0; s >>= 1) { if (tid < s) red[tid] = fmaxf(red[tid], red[tid + s]); __syncthreads(); }
    const float mx = red[0];
    __syncthreads();
    float e = (tid < NC) ? __expf(v - mx) : 0.f;
    red[tid] = e; __syncthreads();
    for (int s = 64; s > 0; s >>= 1) { if (tid < s) red[tid] += red[tid + s]; __syncthreads(); }
    const float denom = red[0];
    if (tid < NC) prob[(size_t)q * NC + tid] = e / denom;
}

// ------------------------------------------------------- WMMA GEMM + fuse ---
// One wave32 per 16(M) x 16(N) output tile.  Two f32 accumulators share the
// same B tile (t), halving B traffic:  accX = x @ t^T,  accS = sigma @ t^T.
//
// fp16 A 16x32 layout: lane l (m = l&15) holds K = kb..kb+7 in v0..3 and
// K = kb+16..kb+23 in v4..7, with kb = (l>>4)*8  -> two 16B loads.
// fp16 B 32x16 layout: lane l (n = l&15) holds K = kb..kb+7 in v0..3 and
// K = kb+16..kb+23 in v4..7, with kb = (l>>4)*8  -> two 16B loads.
// f32 C/D 16x16: VGPR r, lane l -> M = r + 8*(l>>4), N = l&15.
__global__ __launch_bounds__(32)
void gemm_epilogue(const _Float16* __restrict__ x_h,
                   const _Float16* __restrict__ s_h,
                   const _Float16* __restrict__ t_h,
                   const float*    __restrict__ spSum,
                   const float*    __restrict__ sigSum,
                   const float*    __restrict__ tSum,
                   const float*    __restrict__ prob,
                   const int*      __restrict__ tgt_ids,
                   float*          __restrict__ out) {
    const int m0   = blockIdx.x * 16;       // 75 tiles over BQ=1200
    const int n0   = blockIdx.y * 16;       // 13 tiles over NTPAD=208
    const int lane = threadIdx.x;           // wave32
    const int hi   = lane >> 4;             // 0|1
    const int lo   = lane & 15;

    const _Float16* ax = x_h + (size_t)(m0 + lo) * NP + hi * 8;
    const _Float16* as = s_h + (size_t)(m0 + lo) * NP + hi * 8;
    const _Float16* bt = t_h + (size_t)(n0 + lo) * NP + hi * 8;

    v8f accX = {};
    v8f accS = {};

    union V16 { v16h v; v8h h[2]; };

#pragma unroll 2
    for (int k = 0; k < KSTEPS; ++k) {
        __builtin_prefetch(ax + 8 * 32, 0, 1);   // global_prefetch_b8
        __builtin_prefetch(as + 8 * 32, 0, 1);
        __builtin_prefetch(bt + 8 * 32, 0, 1);

        V16 aX, aS, bT;
        aX.h[0] = *(const v8h*)(ax);
        aX.h[1] = *(const v8h*)(ax + 16);
        aS.h[0] = *(const v8h*)(as);
        aS.h[1] = *(const v8h*)(as + 16);
        bT.h[0] = *(const v8h*)(bt);
        bT.h[1] = *(const v8h*)(bt + 16);

        accX = __builtin_amdgcn_wmma_f32_16x16x32_f16(
            false, aX.v, false, bT.v, (short)0, accX, false, false);
        accS = __builtin_amdgcn_wmma_f32_16x16x32_f16(
            false, aS.v, false, bT.v, (short)0, accS, false, false);

        ax += 32; as += 32; bt += 32;
    }

    const float invP = 1.f / (float)NP;
    const int n = n0 + lo;
#pragma unroll
    for (int r = 0; r < 8; ++r) {
        const int m = m0 + r + hi * 8;
        if (n < NT) {
            const float xdot = accX[r];
            const float sdot = accS[r];
            const float cmask = (spSum[m] - xdot) * invP;
            const float cdice = 1.f - (2.f * sdot + 1.f) / (sigSum[m] + tSum[n] + 1.f);
            const float ccls  = -prob[(size_t)m * NC + tgt_ids[n]];
            out[(size_t)m * NT + n] = 2.f * ccls + 5.f * cmask + 5.f * cdice;
        }
    }
}

// ------------------------------------------------------------------ launch --
extern "C" void kernel_launch(void* const* d_in, const int* in_sizes, int n_in,
                              void* d_out, int out_size, void* d_ws, size_t ws_size,
                              hipStream_t stream) {
    (void)in_sizes; (void)n_in; (void)out_size; (void)ws_size;
    const float* pred_logits = (const float*)d_in[0];  // (4,300,80)
    const float* pred_masks  = (const float*)d_in[1];  // (4,300,256,256)
    const int*   tgt_ids     = (const int*)  d_in[2];  // (200,)
    const float* tgt_masks   = (const float*)d_in[3];  // (200,256,256)
    const int*   point_idx   = (const int*)  d_in[4];  // (12544,)
    float* out = (float*)d_out;                        // (4,300,200)

    char*  ws  = (char*)d_ws;
    size_t off = 0;
    auto take = [&](size_t bytes) -> char* {
        char* p = ws + off;
        off = (off + bytes + 255) & ~(size_t)255;
        return p;
    };
    _Float16* t_h    = (_Float16*)take((size_t)NTPAD * NP * sizeof(_Float16)); // 5.2 MB
    _Float16* x_h    = (_Float16*)take((size_t)BQ    * NP * sizeof(_Float16)); // 30.1 MB
    _Float16* s_h    = (_Float16*)take((size_t)BQ    * NP * sizeof(_Float16)); // 30.1 MB
    float*    tSum   = (float*)take(NTPAD * sizeof(float));
    float*    spSum  = (float*)take(BQ * sizeof(float));
    float*    sigSum = (float*)take(BQ * sizeof(float));
    float*    prob   = (float*)take((size_t)BQ * NC * sizeof(float));          // 0.4 MB

    prep_targets<<<NTPAD, 256, 0, stream>>>(tgt_masks, point_idx, t_h, tSum);
    prep_queries<<<BQ, 256, 0, stream>>>(pred_masks, point_idx, x_h, s_h, spSum, sigSum);
    class_prob  <<<BQ, 128, 0, stream>>>(pred_logits, prob);

    dim3 grid(BQ / 16, NTPAD / 16);   // 75 x 13 tiles, one wave32 each
    gemm_epilogue<<<grid, 32, 0, stream>>>(x_h, s_h, t_h, spSum, sigSum, tSum,
                                           prob, tgt_ids, out);
}
